// MSSTBlock_36876589203711
// MI455X (gfx1250) — matrix operand order repaired and checked by reading, not compiled, not run on hardware
//
#include <hip/hip_runtime.h>

#define BB 128
#define CC 64
#define TT 64
#define NN 25
#define KK 8
#define TAU 9
#define DH 16
#define EPSV 1e-5f

typedef __attribute__((ext_vector_type(16))) _Float16 v16h;
typedef __attribute__((ext_vector_type(8)))  _Float16 v8h;
typedef __attribute__((ext_vector_type(8)))  float    v8f;
typedef _Float16 h16;

__device__ __forceinline__ v8f wmma32(v16h a, v16h b, v8f c) {
  // D = A(16x32 f16) x B(32x16 f16) + C(16x16 f32)
  return __builtin_amdgcn_wmma_f32_16x16x32_f16(false, a, false, b, (short)0, c, false, false);
}

// A fragment 16x32 from row-major [row][k]: per lane two contiguous 8-half runs -> 2x b128.
__device__ __forceinline__ v16h load_A(const h16* p, int stride, int klim) {
  int lane = threadIdx.x & 31;
  const h16* pr = p + (lane & 15) * stride;
  int kb = (lane >> 4) * 8;
  v16h a;
#pragma unroll
  for (int j = 0; j < 8; ++j) {
    int k0 = kb + ((j < 4) ? (2 * j) : (16 + 2 * (j - 4)));
    a[2 * j]     = (k0 < klim)     ? pr[k0]     : (h16)0.f;
    a[2 * j + 1] = (k0 + 1 < klim) ? pr[k0 + 1] : (h16)0.f;
  }
  return a;
}

// A fragment gathered from [k][row] storage (transposed source; strided, used only for V).
__device__ __forceinline__ v16h load_At(const h16* p, int stride, int klim) {
  int lane = threadIdx.x & 31;
  const h16* pr = p + (lane & 15);
  int kb = (lane >> 4) * 8;
  v16h a;
#pragma unroll
  for (int j = 0; j < 8; ++j) {
    int k0 = kb + ((j < 4) ? (2 * j) : (16 + 2 * (j - 4)));
    a[2 * j]     = (k0 < klim)     ? pr[k0 * stride]       : (h16)0.f;
    a[2 * j + 1] = (k0 + 1 < klim) ? pr[(k0 + 1) * stride] : (h16)0.f;
  }
  return a;
}

// B fragment 32x16 where source is stored [col][k] row-major: per lane two contiguous 8-half runs -> 2x b128.
__device__ __forceinline__ v16h load_Bt(const h16* p, int stride, int klim) {
  int lane = threadIdx.x & 31;
  const h16* pc = p + (lane & 15) * stride;
  int kb = (lane >> 4) * 8;
  v16h b;
#pragma unroll
  for (int e = 0; e < 16; ++e) {
    int k = kb + ((e < 8) ? e : (8 + e));
    b[e] = (k < klim) ? pc[k] : (h16)0.f;
  }
  return b;
}

// ---------------------------------------------------------------------------
// Kernel 0: S = sum_k D^-1/2 (A_hat^k >= 1) D^-1/2 (symmetric), BN folds,
// f16 weight copies (all kept in natural [out][in] layouts).
// ---------------------------------------------------------------------------
__global__ void prep_kernel(const float* __restrict__ A,
                            const float* __restrict__ sgcn_w, const float* __restrict__ sgcn_b,
                            const float* __restrict__ sgcn_gamma, const float* __restrict__ sgcn_beta,
                            const float* __restrict__ sgcn_mean, const float* __restrict__ sgcn_var,
                            const float* __restrict__ qkv_w, const float* __restrict__ out_w,
                            const float* __restrict__ tconv_w, const float* __restrict__ tconv_b,
                            const float* __restrict__ tbn_gamma, const float* __restrict__ tbn_beta,
                            const float* __restrict__ tbn_mean, const float* __restrict__ tbn_var,
                            h16* __restrict__ S16, h16* __restrict__ Wsg, h16* __restrict__ Wq16,
                            h16* __restrict__ Wo16, h16* __restrict__ Wtc,
                            float* __restrict__ sg_scale, float* __restrict__ sg_bias,
                            float* __restrict__ t_scale, float* __restrict__ t_bias) {
  __shared__ float Ah[NN][NN], Pm[NN][NN], Pn[NN][NN], Sm[NN][NN], dinv[NN];
  int tid = threadIdx.x;
  for (int i = tid; i < NN * NN; i += blockDim.x) {
    int r = i / NN, c = i % NN;
    Ah[r][c] = A[i] + (r == c ? 1.f : 0.f);
    Pm[r][c] = (r == c) ? 1.f : 0.f;
    Sm[r][c] = 0.f;
  }
  __syncthreads();
  for (int k = 0; k <= KK; ++k) {
    if (tid < NN) {
      float s = 0.f;
      for (int m = 0; m < NN; ++m) s += (Pm[tid][m] >= 1.f) ? 1.f : 0.f;
      dinv[tid] = rsqrtf(s);
    }
    __syncthreads();
    if (tid < NN) {
      float di = dinv[tid];
      for (int m = 0; m < NN; ++m)
        Sm[tid][m] += di * ((Pm[tid][m] >= 1.f) ? 1.f : 0.f) * dinv[m];
      for (int m = 0; m < NN; ++m) {
        float acc = 0.f;
        for (int l = 0; l < NN; ++l) acc += Pm[tid][l] * Ah[l][m];
        Pn[tid][m] = acc;
      }
    }
    __syncthreads();
    if (tid < NN) for (int m = 0; m < NN; ++m) Pm[tid][m] = Pn[tid][m];
    __syncthreads();
  }
  for (int i = tid; i < 32 * 32; i += blockDim.x) {
    int r = i >> 5, c = i & 31;
    S16[i] = (h16)((r < NN && c < NN) ? Sm[r][c] : 0.f);
  }
  for (int i = tid; i < CC * CC; i += blockDim.x) Wsg[i] = (h16)sgcn_w[i];       // [o][c]
  for (int i = tid; i < 3 * CC * CC; i += blockDim.x) Wq16[i] = (h16)qkv_w[i];   // [o][c]
  for (int i = tid; i < CC * CC; i += blockDim.x) Wo16[i] = (h16)out_w[i];       // [o][c]
  for (int i = tid; i < TAU * CC * CC; i += blockDim.x) {                        // [dt][o][c]
    int dt = i / (CC * CC); int r = i % (CC * CC); int o = r >> 6; int c = r & 63;
    Wtc[i] = (h16)tconv_w[(o * CC + c) * TAU + dt];
  }
  for (int i = tid; i < CC; i += blockDim.x) {
    float ss = sgcn_gamma[i] * rsqrtf(sgcn_var[i] + EPSV);
    sg_scale[i] = ss;
    sg_bias[i] = (sgcn_b[i] - sgcn_mean[i]) * ss + sgcn_beta[i];
    float ts = tbn_gamma[i] * rsqrtf(tbn_var[i] + EPSV);
    t_scale[i] = ts;
    t_bias[i] = (tconv_b[i] - tbn_mean[i]) * ts + tbn_beta[i];
  }
}

// ---------------------------------------------------------------------------
// Kernel 1: per (b,t): Z = relu(bn(W @ (X @ S))), output xa[b][n][t][c] (f16)
// ---------------------------------------------------------------------------
__global__ void __launch_bounds__(128)
sgcn_kernel(const float* __restrict__ x, const h16* __restrict__ S16,
            const h16* __restrict__ Wsg, const float* __restrict__ sg_scale,
            const float* __restrict__ sg_bias, h16* __restrict__ g_xa) {
  __shared__ h16 Xs[CC][32];   // [c][m]  (A source)
  __shared__ h16 Ys[32][CC];   // [n][c]  (Bt source for stage 2)
  int bt = blockIdx.x, b = bt / TT, t = bt % TT;
  int tid = threadIdx.x;
  for (int i = tid; i < CC * 32; i += 128) {
    int c = i >> 5, n = i & 31;
    Xs[c][n] = (h16)((n < NN) ? x[((b * CC + c) * TT + t) * NN + n] : 0.f);
  }
  __syncthreads();
  int wave = tid >> 5, lane = tid & 31, hl = lane >> 4, ln = lane & 15;
  // stage 1: frag(row=c, col=n) = X[c][m] * S[m][n]; A=Xs rows (mt=wave), B=S^T rows (S symmetric)
  {
    v16h a = load_A(&Xs[wave * 16][0], 32, 32);
#pragma unroll
    for (int nt = 0; nt < 2; ++nt) {
      v16h bf = load_Bt(S16 + (nt * 16) * 32, 32, 32);
      v8f acc = {};
      acc = wmma32(a, bf, acc);
      v8h p;
#pragma unroll
      for (int e = 0; e < 8; ++e) p[e] = (h16)acc[e];
      *(v8h*)&Ys[nt * 16 + ln][wave * 16 + 8 * hl] = p;
    }
  }
  __syncthreads();
  // stage 2: frag(row=o, col=n) = W[o][c] Y[c][n]; A=Wsg (mt=wave), B=load_Bt(Ys[n][c])
  {
    v16h a0 = load_A(Wsg + wave * 16 * CC, CC, 32);
    v16h a1 = load_A(Wsg + wave * 16 * CC + 32, CC, 32);
#pragma unroll
    for (int nt = 0; nt < 2; ++nt) {
      v8f acc = {};
      acc = wmma32(a0, load_Bt(&Ys[nt * 16][0], CC, 32), acc);
      acc = wmma32(a1, load_Bt(&Ys[nt * 16][32], CC, 32), acc);
      int n = nt * 16 + ln;
      if (n < NN) {
        int o0 = wave * 16 + 8 * hl;
        v8h p;
#pragma unroll
        for (int e = 0; e < 8; ++e)
          p[e] = (h16)fmaxf(acc[e] * sg_scale[o0 + e] + sg_bias[o0 + e], 0.f);
        *(v8h*)&g_xa[((size_t)(b * NN + n) * TT + t) * CC + o0] = p;
      }
    }
  }
}

// ---------------------------------------------------------------------------
// Kernel 2: per (b,n): MHA over T=64. All stages in out^T orientation:
// A = weights (row-major), B = activations via load_Bt, packed v8h stores.
// ---------------------------------------------------------------------------
__global__ void __launch_bounds__(128)
attn_kernel(const h16* __restrict__ g_xa, const h16* __restrict__ Wq16,
            const float* __restrict__ qkv_b, const h16* __restrict__ Wo16,
            const float* __restrict__ out_b, h16* __restrict__ att_o) {
  extern __shared__ h16 smem[];
  h16* qkv  = smem;                  // [t][192] = [t][q|k|v]
  h16* att  = smem + 64 * 192;       // [4][t][s]
  h16* ho   = att + 4 * 64 * 64;     // [t][c]
  int bn = blockIdx.x;
  int tid = threadIdx.x, wave = tid >> 5, lane = tid & 31, hl = lane >> 4, ln = lane & 15;
  const h16* xa = g_xa + (size_t)bn * TT * CC;   // [t][c]
  __builtin_prefetch(Wq16 + tid * 64, 0, 0);     // warm L2/WGP$ for weight stream
  // stage A: frag(row=o, col=t); A = Wq16[o][c], B = load_Bt(xa[t][c]); tt fixed = wave
  {
    v16h b0 = load_Bt(xa + (wave * 16) * CC, CC, 32);
    v16h b1 = load_Bt(xa + (wave * 16) * CC + 32, CC, 32);
    int tcol = wave * 16 + ln;
#pragma unroll
    for (int ot = 0; ot < 12; ++ot) {
      v8f acc = {};
      acc = wmma32(load_A(Wq16 + ot * 16 * CC, CC, 32), b0, acc);
      acc = wmma32(load_A(Wq16 + ot * 16 * CC + 32, CC, 32), b1, acc);
      int o0 = ot * 16 + 8 * hl;
      v8h p;
#pragma unroll
      for (int e = 0; e < 8; ++e) p[e] = (h16)(acc[e] + qkv_b[o0 + e]);
      *(v8h*)&qkv[tcol * 192 + o0] = p;
    }
  }
  __syncthreads();
  int h = wave;
  h16* att_h = att + h * TT * TT;
  // stage B: scores^T: frag(row=s, col=t); A = K rows s, B = load_Bt(Q[t][dh]); softmax over s
  {
    v16h ak[4];
#pragma unroll
    for (int st = 0; st < 4; ++st)
      ak[st] = load_A(qkv + (st * 16) * 192 + CC + h * DH, 192, DH);
    for (int tt = 0; tt < 4; ++tt) {
      v16h bq = load_Bt(qkv + (tt * 16) * 192 + h * DH, 192, DH);
      v8f f[4];
#pragma unroll
      for (int st = 0; st < 4; ++st) {
        v8f z = {};
        f[st] = wmma32(ak[st], bq, z);
      }
      float mx = -3.0e38f;
#pragma unroll
      for (int st = 0; st < 4; ++st)
#pragma unroll
        for (int e = 0; e < 8; ++e) mx = fmaxf(mx, f[st][e] * 0.25f);
      mx = fmaxf(mx, __shfl_xor(mx, 16, 32));
      float sum = 0.f;
#pragma unroll
      for (int st = 0; st < 4; ++st)
#pragma unroll
        for (int e = 0; e < 8; ++e) {
          float pv = __expf(f[st][e] * 0.25f - mx);
          f[st][e] = pv;
          sum += pv;
        }
      sum += __shfl_xor(sum, 16, 32);
      float inv = 1.f / sum;
      int tcol = tt * 16 + ln;
#pragma unroll
      for (int st = 0; st < 4; ++st) {
        v8h p;
#pragma unroll
        for (int e = 0; e < 8; ++e) p[e] = (h16)(f[st][e] * inv);
        *(v8h*)&att_h[tcol * 64 + st * 16 + 8 * hl] = p;
      }
    }
  }
  // stage C: O^T: frag(row=j, col=t); A = load_At(V [s][j]), B = load_Bt(att[t][s])
  // (same-wave LDS ops are in-order; no barrier needed between B and C)
  {
    v16h av0 = load_At(qkv + 2 * CC + h * DH, 192, 32);
    v16h av1 = load_At(qkv + 32 * 192 + 2 * CC + h * DH, 192, 32);
#pragma unroll
    for (int tt = 0; tt < 4; ++tt) {
      v8f acc = {};
      acc = wmma32(av0, load_Bt(att_h + (tt * 16) * 64, 64, 32), acc);
      acc = wmma32(av1, load_Bt(att_h + (tt * 16) * 64 + 32, 64, 32), acc);
      int tcol = tt * 16 + ln;
      v8h p;
#pragma unroll
      for (int e = 0; e < 8; ++e) p[e] = (h16)acc[e];
      *(v8h*)&ho[tcol * CC + h * DH + 8 * hl] = p;
    }
  }
  __syncthreads();
  // stage D: out^T: frag(row=o, col=t); A = Wo16[o][c], B = load_Bt(ho[t][c]); tt = wave
  {
    h16* dst = att_o + (size_t)bn * TT * CC;   // [t][c]
    v16h b0 = load_Bt(ho + (wave * 16) * CC, CC, 32);
    v16h b1 = load_Bt(ho + (wave * 16) * CC + 32, CC, 32);
    int tcol = wave * 16 + ln;
#pragma unroll
    for (int ot = 0; ot < 4; ++ot) {
      v8f acc = {};
      acc = wmma32(load_A(Wo16 + ot * 16 * CC, CC, 32), b0, acc);
      acc = wmma32(load_A(Wo16 + ot * 16 * CC + 32, CC, 32), b1, acc);
      int o0 = ot * 16 + 8 * hl;
      v8h p;
#pragma unroll
      for (int e = 0; e < 8; ++e) p[e] = (h16)(acc[e] + out_b[o0 + e]);
      *(v8h*)&dst[tcol * CC + o0] = p;
    }
  }
}

// ---------------------------------------------------------------------------
// Kernel 3: per (b,n): O2 = relu(bn(sum_dt W_dt @ O_shift)) + x
// LDS tile filled with GLOBAL_LOAD_ASYNC_TO_LDS_B128 (ASYNCcnt path, no VGPR
// round-trip); halo rows zeroed with plain ds stores concurrently.
// ---------------------------------------------------------------------------
__global__ void __launch_bounds__(128)
tconv_kernel(const h16* __restrict__ att_o, const h16* __restrict__ Wtc,
             const float* __restrict__ t_scale, const float* __restrict__ t_bias,
             const float* __restrict__ x, float* __restrict__ out) {
  __shared__ h16 Os[TT + 8][CC];   // [4 + t][c], halo of 4 zero rows each side
  int bn = blockIdx.x, b = bn / NN, n = bn % NN;
  int tid = threadIdx.x, wave = tid >> 5, lane = tid & 31, hl = lane >> 4, ln = lane & 15;
  const h16* src = att_o + (size_t)bn * TT * CC;   // [t][c]
  __builtin_prefetch(Wtc + tid * 64, 0, 0);        // warm L2/WGP$ for weight stream
  // interior rows: async DMA global -> LDS, 16 bytes per lane per issue
  for (int i = tid; i < TT * (CC / 8); i += 128) {
    int t = i >> 3, cb = (i & 7) * 8;
    unsigned lds = (unsigned)(uintptr_t)&Os[t + 4][cb];
    const h16* gp = src + t * CC + cb;
    asm volatile("global_load_async_to_lds_b128 %0, %1, off"
                 :: "v"(lds), "v"(gp) : "memory");
  }
  // halo rows: zero-fill
  for (int i = tid; i < 8 * (CC / 8); i += 128) {
    int r = i >> 3, cb = (i & 7) * 8;
    int tt = (r < 4) ? r : (TT + r);
    v8h z = {};
    *(v8h*)&Os[tt][cb] = z;
  }
  asm volatile("s_wait_asynccnt 0x0" ::: "memory");
  __syncthreads();
  // wave owns time-tile nt = wave; sweep 4 out-channel tiles with hoisted B frags per shift
  int nt = wave;
  v8f acc[4] = {};
  for (int dt = 0; dt < TAU; ++dt) {
    v16h b0 = load_Bt(&Os[nt * 16 + dt][0], CC, 32);
    v16h b1 = load_Bt(&Os[nt * 16 + dt][32], CC, 32);
#pragma unroll
    for (int ot = 0; ot < 4; ++ot) {
      acc[ot] = wmma32(load_A(Wtc + dt * CC * CC + ot * 16 * CC, CC, 32), b0, acc[ot]);
      acc[ot] = wmma32(load_A(Wtc + dt * CC * CC + ot * 16 * CC + 32, CC, 32), b1, acc[ot]);
    }
  }
  int t = nt * 16 + ln;
#pragma unroll
  for (int ot = 0; ot < 4; ++ot) {
#pragma unroll
    for (int e = 0; e < 8; ++e) {
      int oc = ot * 16 + 8 * hl + e;
      float v = fmaxf(acc[ot][e] * t_scale[oc] + t_bias[oc], 0.f);
      size_t idx = ((size_t)(b * CC + oc) * TT + t) * NN + n;
      out[idx] = v + x[idx];
    }
  }
}

// ---------------------------------------------------------------------------
extern "C" void kernel_launch(void* const* d_in, const int* in_sizes, int n_in,
                              void* d_out, int out_size, void* d_ws, size_t ws_size,
                              hipStream_t stream) {
  const float* x       = (const float*)d_in[0];
  const float* A       = (const float*)d_in[1];
  const float* sgcn_w  = (const float*)d_in[2];
  const float* sgcn_b  = (const float*)d_in[3];
  const float* sgcn_g  = (const float*)d_in[4];
  const float* sgcn_be = (const float*)d_in[5];
  const float* sgcn_m  = (const float*)d_in[6];
  const float* sgcn_v  = (const float*)d_in[7];
  const float* qkv_w   = (const float*)d_in[8];
  const float* qkv_b   = (const float*)d_in[9];
  const float* out_w   = (const float*)d_in[10];
  const float* out_b   = (const float*)d_in[11];
  const float* tconv_w = (const float*)d_in[12];
  const float* tconv_b = (const float*)d_in[13];
  const float* tbn_g   = (const float*)d_in[14];
  const float* tbn_be  = (const float*)d_in[15];
  const float* tbn_m   = (const float*)d_in[16];
  const float* tbn_v   = (const float*)d_in[17];

  char* ws = (char*)d_ws;
  size_t off = 0;
  auto take = [&](size_t bytes) -> char* {
    char* p = ws + off;
    off = (off + bytes + 255) & ~(size_t)255;
    return p;
  };
  h16* S16  = (h16*)take(32 * 32 * 2);
  h16* Wsg  = (h16*)take(CC * CC * 2);
  h16* Wq16 = (h16*)take(3 * CC * CC * 2);
  h16* Wo16 = (h16*)take(CC * CC * 2);
  h16* Wt16 = (h16*)take(TAU * CC * CC * 2);
  float* sgs = (float*)take(CC * 4);
  float* sgb = (float*)take(CC * 4);
  float* tsc = (float*)take(CC * 4);
  float* tbi = (float*)take(CC * 4);
  h16* g_xa = (h16*)take((size_t)BB * NN * TT * CC * 2);   // ~26 MB, [b][n][t][c]
  h16* ao   = (h16*)take((size_t)BB * NN * TT * CC * 2);   // ~26 MB, [b][n][t][c]

  prep_kernel<<<1, 256, 0, stream>>>(A, sgcn_w, sgcn_b, sgcn_g, sgcn_be, sgcn_m, sgcn_v,
                                     qkv_w, out_w, tconv_w, tconv_b, tbn_g, tbn_be, tbn_m, tbn_v,
                                     S16, Wsg, Wq16, Wo16, Wt16, sgs, sgb, tsc, tbi);
  sgcn_kernel<<<BB * TT, 128, 0, stream>>>(x, S16, Wsg, sgs, sgb, g_xa);
  attn_kernel<<<BB * NN, 128, (64 * 192 + 4 * 64 * 64 + 64 * 64) * sizeof(h16), stream>>>(
      g_xa, Wq16, qkv_b, Wo16, out_b, ao);
  tconv_kernel<<<BB * NN, 128, 0, stream>>>(ao, Wt16, tsc, tbi, x, (float*)d_out);
  (void)in_sizes; (void)n_in; (void)out_size; (void)ws_size;
}
